// NerfactoField_55465207661106
// MI455X (gfx1250) — matrix operand order repaired and checked by reading, not compiled, not run on hardware
//
#include <hip/hip_runtime.h>
#include <cmath>

#define NLEV   16
#define MAXT   (1u << 19)   // max table size per level
#define WAVES  8
#define TILE   16
#define TPW    8            // tiles per wave (amortizes weight staging)
#define NTHR   (WAVES * 32)

typedef _Float16 v16h __attribute__((ext_vector_type(16)));
typedef _Float16 v8h  __attribute__((ext_vector_type(8)));
typedef float    v8f  __attribute__((ext_vector_type(8)));

struct HashParams { int res[NLEV]; };

// bias offsets inside sBias[]
#define DIRB0 0
#define DIRB1 16
#define DENB0 32
#define DENB1 96
#define DENB2 160
#define COLB0 176
#define COLB1 240
#define COLB2 304
#define COLB3 368

// ---- WMMA fragment loaders (f16, 16x16x32) ------------------------------
// A (16xK row-major in LDS): lane half 0 -> K {0..7,16..23}, half 1 -> K {8..15,24..31}
__device__ __forceinline__ v16h ld_a(const _Float16* buf, int stride, int kbase, int lane) {
  const int m = lane & 15, kh = lane >> 4;
  const _Float16* p = buf + m * stride + kbase + kh * 8;
  v8h lo = *(const v8h*)(p);
  v8h hi = *(const v8h*)(p + 16);
  v16h r;
#pragma unroll
  for (int i = 0; i < 8; ++i) { r[i] = lo[i]; r[i + 8] = hi[i]; }
  return r;
}

// B (KxN), weights stored transposed [N][Kpad] in LDS:
// lane half 0 -> K {0..7,16..23}, half 1 -> K {8..15,24..31}
__device__ __forceinline__ v16h ld_b(const _Float16* wt, int kpad, int ntile, int kbase, int lane) {
  const int n = lane & 15, kh = lane >> 4;
  const _Float16* p = wt + (ntile * 16 + n) * kpad + kbase + kh * 8;
  v8h lo = *(const v8h*)(p);
  v8h hi = *(const v8h*)(p + 16);
  v16h r;
#pragma unroll
  for (int i = 0; i < 8; ++i) { r[i] = lo[i]; r[i + 8] = hi[i]; }
  return r;
}

// D epilogue: bias + (optional) relu, store f16 activation tile row-major.
// C/D layout: VGPR j -> row j (lanes 0-15) / row j+8 (lanes 16-31), col = lane&15
__device__ __forceinline__ void st_d(_Float16* dst, int stride, int nbase, v8f acc,
                                     const float* bias, bool relu, int lane) {
  const int n = lane & 15, mh = lane >> 4;
  const float bv = bias[nbase + n];
#pragma unroll
  for (int j = 0; j < 8; ++j) {
    float v = acc[j] + bv;
    if (relu) v = fmaxf(v, 0.0f);
    dst[(j + mh * 8) * stride + nbase + n] = (_Float16)v;
  }
}

__device__ __forceinline__ v8f mlp_acc(const _Float16* in, int strideIn, int kchunks,
                                       const _Float16* wt, int kpad, int ntile, int lane) {
  v8f acc = {};
  v16h a0 = ld_a(in, strideIn, 0, lane);
  v16h b0 = ld_b(wt, kpad, ntile, 0, lane);
  acc = __builtin_amdgcn_wmma_f32_16x16x32_f16(false, a0, false, b0, (short)0, acc, false, false);
  if (kchunks > 1) {
    v16h a1 = ld_a(in, strideIn, 32, lane);
    v16h b1 = ld_b(wt, kpad, ntile, 32, lane);
    acc = __builtin_amdgcn_wmma_f32_16x16x32_f16(false, a1, false, b1, (short)0, acc, false, false);
  }
  return acc;
}

__device__ __forceinline__ void mlp_layer(const _Float16* in, int strideIn, int kchunks,
                                          const _Float16* wt, int kpad,
                                          _Float16* outb, int strideOut, int ntiles,
                                          const float* bias, bool relu, int lane) {
  v16h a0 = ld_a(in, strideIn, 0, lane);
  v16h a1 = a0;
  if (kchunks > 1) a1 = ld_a(in, strideIn, 32, lane);
  for (int nt = 0; nt < ntiles; ++nt) {
    v8f acc = {};
    v16h b0 = ld_b(wt, kpad, nt, 0, lane);
    acc = __builtin_amdgcn_wmma_f32_16x16x32_f16(false, a0, false, b0, (short)0, acc, false, false);
    if (kchunks > 1) {
      v16h b1 = ld_b(wt, kpad, nt, 32, lane);
      acc = __builtin_amdgcn_wmma_f32_16x16x32_f16(false, a1, false, b1, (short)0, acc, false, false);
    }
    st_d(outb, strideOut, nt * 16, acc, bias, relu, lane);
  }
}

// cooperative f32 -> f16 transposed weight staging: dst[n][k] = src[k][n]
__device__ __forceinline__ void load_wt(_Float16* dst, const float* src,
                                        int Nw, int Kpad, int Kreal, int ncols, int nreal, int tid) {
  const int tot = Nw * Kpad;
  for (int idx = tid; idx < tot; idx += NTHR) {
    const int n = idx / Kpad;
    const int k = idx - n * Kpad;
    float v = 0.0f;
    if (k < Kreal && n < nreal) v = src[k * ncols + n];
    dst[idx] = (_Float16)v;
  }
}

__global__ __launch_bounds__(NTHR, 1)
void nerfacto_wmma_kernel(
    const float* __restrict__ ray, const float* __restrict__ dirs,
    const int* __restrict__ cam, const float* __restrict__ aabb,
    const float* __restrict__ hashT,
    const float* __restrict__ dw0, const float* __restrict__ db0,
    const float* __restrict__ dw1, const float* __restrict__ db1,
    const float* __restrict__ nw0, const float* __restrict__ nb0,
    const float* __restrict__ nw1, const float* __restrict__ nb1,
    const float* __restrict__ nw2, const float* __restrict__ nb2,
    const float* __restrict__ cw0, const float* __restrict__ cb0,
    const float* __restrict__ cw1, const float* __restrict__ cb1,
    const float* __restrict__ cw2, const float* __restrict__ cb2,
    const float* __restrict__ cw3, const float* __restrict__ cb3,
    const float* __restrict__ appE,
    float* __restrict__ out, int nSamples, HashParams P)
{
  __shared__ __align__(16) _Float16 sDirW0[16 * 32], sDirW1[16 * 32];
  __shared__ __align__(16) _Float16 sDenW0[64 * 32], sDenW1[64 * 64], sDenW2[16 * 64];
  __shared__ __align__(16) _Float16 sColW0[64 * 64], sColW1[64 * 64], sColW2[64 * 64], sColW3[16 * 64];
  __shared__ float sBias[384];
  __shared__ int   sRes[NLEV];
  __shared__ __align__(16) _Float16 sAct[WAVES][3][16 * 64];  // per-wave ping/pong/ci
  __shared__ __align__(16) float sPos[WAVES][16][4];          // pos.xyz + cam idx

  const int tid  = threadIdx.x;
  const int lane = tid & 31;
  const int wv   = tid >> 5;

  // ---- stage weights once per block (transposed, f16, zero-padded) ----
  load_wt(sDirW0, dw0, 16, 32, 3, 16, 16, tid);
  load_wt(sDirW1, dw1, 16, 32, 16, 16, 16, tid);
  load_wt(sDenW0, nw0, 64, 32, 32, 64, 64, tid);
  load_wt(sDenW1, nw1, 64, 64, 64, 64, 64, tid);
  load_wt(sDenW2, nw2, 16, 64, 64, 16, 16, tid);
  // col_w0: reference ci = [geo(15) | d(16) | app(32)]; our ci = [d | app | geo | 0]
  for (int idx = tid; idx < 64 * 64; idx += NTHR) {
    const int nn = idx >> 6;
    const int k  = idx & 63;
    const int kRef = (k < 48) ? (k + 15) : ((k < 63) ? (k - 48) : -1);
    sColW0[idx] = (_Float16)((kRef >= 0) ? cw0[kRef * 64 + nn] : 0.0f);
  }
  load_wt(sColW1, cw1, 64, 64, 64, 64, 64, tid);
  load_wt(sColW2, cw2, 64, 64, 64, 64, 64, tid);
  load_wt(sColW3, cw3, 16, 64, 64, 3, 3, tid);
  if (tid < 16) {
    sBias[DIRB0 + tid] = db0[tid];
    sBias[DIRB1 + tid] = db1[tid];
    sBias[DENB2 + tid] = nb2[tid];
    sBias[COLB3 + tid] = (tid < 3) ? cb3[tid] : 0.0f;
  }
  if (tid < 64) {
    sBias[DENB0 + tid] = nb0[tid];
    sBias[DENB1 + tid] = nb1[tid];
    sBias[COLB0 + tid] = cb0[tid];
    sBias[COLB1 + tid] = cb1[tid];
    sBias[COLB2 + tid] = cb2[tid];
  }
  if (tid < NLEV) sRes[tid] = P.res[tid];
  __syncthreads();

  _Float16* bufA = sAct[wv][0];
  _Float16* bufB = sAct[wv][1];
  _Float16* bufC = sAct[wv][2];   // ci tile: [d(0..15) | app(16..47) | geo(48..62) | 0]

  // zero everything once (establishes bufC col 63 == 0 permanently)
  {
    v8h z = {};
    v8h* pz = (v8h*)sAct[wv][0];
    for (int i = lane; i < (3 * 16 * 64) / 8; i += 32) pz[i] = z;
  }

  const int tile0 = (blockIdx.x * WAVES + wv) * TPW;

#pragma unroll 1
  for (int t = 0; t < TPW; ++t) {
    const int base = (tile0 + t) * TILE;
    if (base >= nSamples) break;

    // re-zero ping/pong buffers: dir-input padding (bufB cols 3..31) and
    // dir-hidden padding (bufA cols 16..31) must be zero each tile
    {
      v8h z = {};
      v8h* pz = (v8h*)sAct[wv][0];
      for (int i = lane; i < (2 * 16 * 64) / 8; i += 32) pz[i] = z;
    }

    // per-sample scalar prep (one sample per lane 0..15)
    if (lane < TILE) {
      const int s = base + lane;
      const float a0 = aabb[0], a1 = aabb[1], a2 = aabb[2];
      const float b0 = aabb[3], b1 = aabb[4], b2 = aabb[5];
      float px = (ray[(size_t)s * 3 + 0] - a0) / (b0 - a0);
      float py = (ray[(size_t)s * 3 + 1] - a1) / (b1 - a1);
      float pz = (ray[(size_t)s * 3 + 2] - a2) / (b2 - a2);
      sPos[wv][lane][0] = fminf(fmaxf(px, 0.0f), 1.0f);
      sPos[wv][lane][1] = fminf(fmaxf(py, 0.0f), 1.0f);
      sPos[wv][lane][2] = fminf(fmaxf(pz, 0.0f), 1.0f);
      sPos[wv][lane][3] = __int_as_float(cam[s]);
      bufB[lane * 64 + 0] = (_Float16)dirs[(size_t)s * 3 + 0];
      bufB[lane * 64 + 1] = (_Float16)dirs[(size_t)s * 3 + 1];
      bufB[lane * 64 + 2] = (_Float16)dirs[(size_t)s * 3 + 2];
    }

    // ---- direction MLP: 3(pad32)->16 relu ->16, result into ci cols 0..15 ----
    mlp_layer(bufB, 64, 1, sDirW0, 32, bufA, 64, 1, sBias + DIRB0, true, lane);
    mlp_layer(bufA, 64, 1, sDirW1, 32, bufC, 64, 1, sBias + DIRB1, false, lane);

    // ---- hash encoding: 16 levels x 16 samples = 256 gathers -> bufA cols 0..31 ----
#pragma unroll
    for (int it = 0; it < 8; ++it) {
      const int p = it * 32 + lane;
      const int lev = p >> 4, samp = p & 15;
      const float4 pp = *(const float4*)(&sPos[wv][samp][0]);
      const float fr = (float)(sRes[lev] - 1);
      const unsigned cx = (unsigned)fminf(fmaxf(floorf(pp.x * fr), 0.0f), fr);
      const unsigned cy = (unsigned)fminf(fmaxf(floorf(pp.y * fr), 0.0f), fr);
      const unsigned cz = (unsigned)fminf(fmaxf(floorf(pp.z * fr), 0.0f), fr);
      const unsigned h = (cx + cy * 2654435761u + cz * 805459861u) & 4095u; // % HASH_MOD(4096)
      const float2 f = *(const float2*)(hashT + ((size_t)lev * MAXT + (size_t)h) * 2);
      union { _Float16 h2[2]; unsigned u; } pk;
      pk.h2[0] = (_Float16)f.x; pk.h2[1] = (_Float16)f.y;
      *(unsigned*)(bufA + samp * 64 + lev * 2) = pk.u;
    }

    // ---- appearance embedding gather -> ci cols 16..47 ----
#pragma unroll
    for (int it = 0; it < 8; ++it) {
      const int p = it * 32 + lane;
      const int samp = p >> 4, pr = p & 15;
      const int ci = __float_as_int(sPos[wv][samp][3]);
      const float2 f = *(const float2*)(appE + (size_t)ci * 32 + pr * 2);
      union { _Float16 h2[2]; unsigned u; } pk;
      pk.h2[0] = (_Float16)f.x; pk.h2[1] = (_Float16)f.y;
      *(unsigned*)(bufC + samp * 64 + 16 + pr * 2) = pk.u;
    }

    // ---- density MLP: 32->64 relu ->64 relu ->16 ----
    mlp_layer(bufA, 64, 1, sDenW0, 32, bufB, 64, 4, sBias + DENB0, true, lane);
    mlp_layer(bufB, 64, 2, sDenW1, 64, bufA, 64, 4, sBias + DENB1, true, lane);
    {
      v8f df = mlp_acc(bufA, 64, 2, sDenW2, 64, 0, lane);
      const int n = lane & 15, mh = lane >> 4;
      const float bv = sBias[DENB2 + n];
#pragma unroll
      for (int j = 0; j < 8; ++j) {
        const float v = df[j] + bv;
        const int m = j + mh * 8;
        const size_t orow = (size_t)(base + m) * 19;
        if (n == 0) {
          out[orow + 3] = fmaxf(v, 0.0f);            // density
        } else {
          out[orow + 4 + (n - 1)] = v;               // geo passthrough
          bufC[m * 64 + 48 + (n - 1)] = (_Float16)v; // geo into ci
        }
      }
    }

    // ---- color MLP: 63(pad64)->64 relu ->64 relu ->64 relu ->3 sigmoid ----
    mlp_layer(bufC, 64, 2, sColW0, 64, bufA, 64, 4, sBias + COLB0, true, lane);
    mlp_layer(bufA, 64, 2, sColW1, 64, bufB, 64, 4, sBias + COLB1, true, lane);
    mlp_layer(bufB, 64, 2, sColW2, 64, bufA, 64, 4, sBias + COLB2, true, lane);
    {
      v8f cf = mlp_acc(bufA, 64, 2, sColW3, 64, 0, lane);
      const int n = lane & 15, mh = lane >> 4;
      if (n < 3) {
        const float bv = sBias[COLB3 + n];
#pragma unroll
        for (int j = 0; j < 8; ++j) {
          const float v = cf[j] + bv;
          const int m = j + mh * 8;
          out[(size_t)(base + m) * 19 + n] = 1.0f / (1.0f + expf(-v));
        }
      }
    }
  }
}

extern "C" void kernel_launch(void* const* d_in, const int* in_sizes, int n_in,
                              void* d_out, int out_size, void* d_ws, size_t ws_size,
                              hipStream_t stream) {
  (void)n_in; (void)out_size; (void)d_ws; (void)ws_size;
  const float* ray   = (const float*)d_in[0];
  const float* dirs  = (const float*)d_in[1];
  const int*   cam   = (const int*)d_in[2];
  const float* aabb  = (const float*)d_in[3];
  const float* hashT = (const float*)d_in[4];
  const float* dw0 = (const float*)d_in[5];
  const float* db0 = (const float*)d_in[6];
  const float* dw1 = (const float*)d_in[7];
  const float* db1 = (const float*)d_in[8];
  const float* nw0 = (const float*)d_in[9];
  const float* nb0 = (const float*)d_in[10];
  const float* nw1 = (const float*)d_in[11];
  const float* nb1 = (const float*)d_in[12];
  const float* nw2 = (const float*)d_in[13];
  const float* nb2 = (const float*)d_in[14];
  const float* cw0 = (const float*)d_in[15];
  const float* cb0 = (const float*)d_in[16];
  const float* cw1 = (const float*)d_in[17];
  const float* cb1 = (const float*)d_in[18];
  const float* cw2 = (const float*)d_in[19];
  const float* cb2 = (const float*)d_in[20];
  const float* cw3 = (const float*)d_in[21];
  const float* cb3 = (const float*)d_in[22];
  const float* appE = (const float*)d_in[23];
  float* out = (float*)d_out;

  const int n = in_sizes[0] / 3;

  HashParams P;
  const double growth = exp((log(2048.0) - log(16.0)) / 15.0);
  for (int i = 0; i < NLEV; ++i) P.res[i] = (int)(16.0 * pow(growth, (double)i));

  const int samplesPerBlock = WAVES * TILE * TPW;
  const int blocks = (n + samplesPerBlock - 1) / samplesPerBlock;
  nerfacto_wmma_kernel<<<blocks, NTHR, 0, stream>>>(
      ray, dirs, cam, aabb, hashT,
      dw0, db0, dw1, db1,
      nw0, nb0, nw1, nb1, nw2, nb2,
      cw0, cb0, cw1, cb1, cw2, cb2, cw3, cb3,
      appE, out, n, P);
}
